// SelfAttention_37486474559894
// MI455X (gfx1250) — matrix operand order repaired and checked by reading, not compiled, not run on hardware
//
#include <hip/hip_runtime.h>
#include <hip/hip_bf16.h>

// ---------------------------------------------------------------------------
// Self-attention (S=4096, DIM=1024, 16 heads x 64) for gfx1250 / MI455X.
// WMMA f16 everywhere; flash-attention stage 2 with:
//   - ones-column WMMA row-sum (no shuffle reductions for the denominator)
//   - 64-key tiles (halves max-shuffle + rescale rate)
//   - K/V tiles staged in LDS, shared by 4 waves, double buffered via
//     global_load_async_to_lds_b128 (ASYNCcnt) with load+ds_store fallback
// ---------------------------------------------------------------------------

#define S_LEN 4096
#define DIM   1024
#define NH    16
#define HD    64
#define KP    72   // padded LDS row pitch in halves (144B = 16B-aligned, 36-bank stagger)

#ifndef TRY_ASYNC
#define TRY_ASYNC 1
#endif
#if TRY_ASYNC && __has_builtin(__builtin_amdgcn_global_load_async_to_lds_b128)
#define HAVE_ASYNC 1
#else
#define HAVE_ASYNC 0
#endif

typedef __attribute__((ext_vector_type(16))) _Float16 v16h;
typedef __attribute__((ext_vector_type(8)))  _Float16 v8h;
typedef __attribute__((ext_vector_type(4)))  _Float16 v4h;
typedef __attribute__((ext_vector_type(8)))  float    v8f;

#if HAVE_ASYNC
typedef int i4 __attribute__((vector_size(16)));
typedef __attribute__((address_space(1))) i4 gi4;
typedef __attribute__((address_space(3))) i4 li4;
#endif

__device__ __forceinline__ v16h cat8(v8h lo, v8h hi) {
    return __builtin_shufflevector(lo, hi, 0,1,2,3,4,5,6,7,8,9,10,11,12,13,14,15);
}

// B fragment (32x16 f16): per lane 16 contiguous halves (lane<16: K=0..15, else 16..31)
__device__ __forceinline__ v16h load_b16(const _Float16* p) {
    v8h lo = *(const v8h*)(p);
    v8h hi = *(const v8h*)(p + 8);
    return cat8(lo, hi);
}

// A fragment (16x32 f16): lanes0-15 K=0..15; lanes16-31 K=16..31
__device__ __forceinline__ v16h load_a16(const _Float16* p, int hi16) {
    v8h lo = *(const v8h*)(p + hi16 * 16);
    v8h hh = *(const v8h*)(p + 8 + hi16 * 16);
    return cat8(lo, hh);
}

__device__ __forceinline__ v8f wmma16(v16h a, v16h b, v8f c) {
    return __builtin_amdgcn_wmma_f32_16x16x32_f16(false, a, false, b, (short)0, c,
                                                  false, false);
}

// Stage a 64-row x 64-half tile (global pitch `pitch`, LDS pitch KP).
// 128 threads x 4 chunks of 16B = 64 rows x 128B.
__device__ __forceinline__ void stage_tile(_Float16* dst, const _Float16* src,
                                           int pitch, int tid) {
#pragma unroll
    for (int i = 0; i < 4; ++i) {
        const int c = tid * 4 + i;
        const int row = c >> 3;
        const int col = (c & 7) * 8;
        const _Float16* g = src + (size_t)row * pitch + col;
        _Float16* l = dst + row * KP + col;
#if HAVE_ASYNC
        __builtin_amdgcn_global_load_async_to_lds_b128(
            (gi4*)(unsigned long long)(const void*)g,
            (li4*)(unsigned)(unsigned long long)(void*)l,
            0, 0);
#else
        *(v8h*)l = *(const v8h*)g;
#endif
    }
}

__device__ __forceinline__ void wait_async() {
#if HAVE_ASYNC
#if __has_builtin(__builtin_amdgcn_s_wait_asynccnt)
    __builtin_amdgcn_s_wait_asynccnt(0);
#else
    asm volatile("s_wait_asynccnt 0" ::: "memory");
#endif
#endif
}

// --------------------------- conversion kernels ----------------------------

__global__ __launch_bounds__(256) void cvt_x_f16(const float* __restrict__ src,
                                                 _Float16* __restrict__ dst, int n4) {
    int i = blockIdx.x * 256 + threadIdx.x;
    if (i < n4) {
        float4 v = ((const float4*)src)[i];
        v4h h = { (_Float16)v.x, (_Float16)v.y, (_Float16)v.z, (_Float16)v.w };
        ((v4h*)dst)[i] = h;
    }
}

// W[k][n] (f32) -> Wt[n][k] (f16); k fastest => coalesced writes.
__global__ __launch_bounds__(256) void cvt_w_t(const float* __restrict__ W,
                                               _Float16* __restrict__ Wt) {
    int k = blockIdx.x * 256 + threadIdx.x;
    int n = blockIdx.y;
    Wt[(size_t)n * DIM + k] = (_Float16)W[(size_t)k * DIM + n];
}

// ------------------------------ QKV GEMM -----------------------------------
// Each wave: 32 rows x 64 cols (2 A-frags share every B-frag).
// trans!=0 -> store [DIM][S] (for V). oscale folds softmax scaling into Q.

__global__ __launch_bounds__(128) void gemm_bias_f16(
    const _Float16* __restrict__ A,   // [S, DIM] f16
    const _Float16* __restrict__ Bt,  // [DIM(n), DIM(k)] f16 transposed weight
    const float*    __restrict__ bias,
    _Float16*       __restrict__ Dst,
    float oscale, int trans)
{
    const int lane = threadIdx.x;
    const int lh = lane & 15;
    const int hi = lane >> 4;
    const int m0 = (blockIdx.x * 4 + threadIdx.y) * 32;
    const int n0 = blockIdx.y * 64;

    v8f acc[8];
#pragma unroll
    for (int i = 0; i < 8; ++i) acc[i] = (v8f){0,0,0,0,0,0,0,0};

    const _Float16* A0 = A + (size_t)(m0 + lh) * DIM;
    const _Float16* A1 = A + (size_t)(m0 + 16 + lh) * DIM;
    const _Float16* B0 = Bt + (size_t)(n0 +  0 + lh) * DIM + hi * 16;
    const _Float16* B1 = Bt + (size_t)(n0 + 16 + lh) * DIM + hi * 16;
    const _Float16* B2 = Bt + (size_t)(n0 + 32 + lh) * DIM + hi * 16;
    const _Float16* B3 = Bt + (size_t)(n0 + 48 + lh) * DIM + hi * 16;

    for (int k = 0; k < DIM; k += 32) {
        __builtin_prefetch(A0 + k + 512, 0, 1);   // global_prefetch_b8
        __builtin_prefetch(B0 + k + 512, 0, 1);
        v16h a0 = load_a16(A0 + k, hi);
        v16h a1 = load_a16(A1 + k, hi);
        v16h b;
        b = load_b16(B0 + k); acc[0] = wmma16(a0, b, acc[0]); acc[4] = wmma16(a1, b, acc[4]);
        b = load_b16(B1 + k); acc[1] = wmma16(a0, b, acc[1]); acc[5] = wmma16(a1, b, acc[5]);
        b = load_b16(B2 + k); acc[2] = wmma16(a0, b, acc[2]); acc[6] = wmma16(a1, b, acc[6]);
        b = load_b16(B3 + k); acc[3] = wmma16(a0, b, acc[3]); acc[7] = wmma16(a1, b, acc[7]);
    }

#pragma unroll
    for (int half = 0; half < 2; ++half) {
#pragma unroll
        for (int c = 0; c < 4; ++c) {
            const v8f& ac = acc[half * 4 + c];
            const int n = n0 + c * 16 + lh;
            const float bs = bias[n];
            if (!trans) {
#pragma unroll
                for (int j = 0; j < 8; ++j) {
                    const int m = m0 + half * 16 + j + hi * 8;
                    Dst[(size_t)m * DIM + n] = (_Float16)((ac[j] + bs) * oscale);
                }
            } else {
                v8h pack;
#pragma unroll
                for (int j = 0; j < 8; ++j) pack[j] = (_Float16)((ac[j] + bs) * oscale);
                *(v8h*)(Dst + (size_t)n * S_LEN + m0 + half * 16 + hi * 8) = pack;
            }
        }
    }
}

// --------------------------- flash attention -------------------------------
// Block = 4 waves, one head, 4 query tiles of 16. 64-key steps.
// K/V tiles staged in LDS (double buffered), P transposed through LDS,
// softmax denominator accumulated by a ones-column WMMA.

__global__ __launch_bounds__(128) void attn_fa(
    const _Float16* __restrict__ Qh,  // [S][DIM] f16, pre-scaled by 1/8*log2(e)
    const _Float16* __restrict__ Kh,  // [S][DIM] f16
    const _Float16* __restrict__ Vt,  // [DIM][S] f16
    float*          __restrict__ Out) // [S][DIM] f32
{
    __shared__ __align__(16) _Float16 sK[2][64 * KP];
    __shared__ __align__(16) _Float16 sV[2][64 * KP];
    __shared__ __align__(16) _Float16 sP[4][16 * KP];

    const int lane = threadIdx.x;
    const int wv = threadIdx.y;
    const int tid = wv * 32 + lane;
    const int lh = lane & 15;
    const int hi = lane >> 4;
    const int head = blockIdx.y;
    const int q0 = (blockIdx.x * 4 + wv) * 16;

    const _Float16* Qrow = Qh + (size_t)(q0 + lh) * DIM + head * HD;
    const v16h qa0 = load_a16(Qrow, hi);        // dims 0..31
    const v16h qa1 = load_a16(Qrow + 32, hi);   // dims 32..63

    v16h vones;
#pragma unroll
    for (int i = 0; i < 16; ++i) vones[i] = (_Float16)1.0f;

    v8f o0 = {0,0,0,0,0,0,0,0};
    v8f o1 = o0, o2 = o0, o3 = o0, o4 = o0;   // o4 = running softmax denominator
    float mrun[8];
#pragma unroll
    for (int j = 0; j < 8; ++j) mrun[j] = -1e30f;

    _Float16* P = sP[wv];
    const _Float16* Kg = Kh + head * HD;                   // + (kb+row)*DIM
    const _Float16* Vg = Vt + (size_t)head * HD * S_LEN;   // + row*S_LEN + kb

    stage_tile(&sK[0][0], Kg, DIM, tid);
    stage_tile(&sV[0][0], Vg, S_LEN, tid);
    wait_async();
    __syncthreads();

    int buf = 0;
    for (int kb = 0; kb < S_LEN; kb += 64, buf ^= 1) {
        const _Float16* kt = &sK[buf][0];
        const _Float16* vt = &sV[buf][0];

        // ---- scores: 4 n-halves x 2 k-chunks -----------------------------
        v8f s0 = {0,0,0,0,0,0,0,0};
        v8f s1 = s0, s2 = s0, s3 = s0;
        s0 = wmma16(qa0, load_b16(kt + (0 * 16 + lh) * KP +  0 + hi * 16), s0);
        s0 = wmma16(qa1, load_b16(kt + (0 * 16 + lh) * KP + 32 + hi * 16), s0);
        s1 = wmma16(qa0, load_b16(kt + (1 * 16 + lh) * KP +  0 + hi * 16), s1);
        s1 = wmma16(qa1, load_b16(kt + (1 * 16 + lh) * KP + 32 + hi * 16), s1);
        s2 = wmma16(qa0, load_b16(kt + (2 * 16 + lh) * KP +  0 + hi * 16), s2);
        s2 = wmma16(qa1, load_b16(kt + (2 * 16 + lh) * KP + 32 + hi * 16), s2);
        s3 = wmma16(qa0, load_b16(kt + (3 * 16 + lh) * KP +  0 + hi * 16), s3);
        s3 = wmma16(qa1, load_b16(kt + (3 * 16 + lh) * KP + 32 + hi * 16), s3);

        // ---- overlap: kick off next tile's staging -----------------------
        if (kb + 64 < S_LEN) {
            stage_tile(&sK[buf ^ 1][0], Kg + (size_t)(kb + 64) * DIM, DIM, tid);
            stage_tile(&sV[buf ^ 1][0], Vg + (kb + 64), S_LEN, tid);
        }

        // ---- online softmax (logits already in exp2 domain) --------------
#pragma unroll
        for (int j = 0; j < 8; ++j) {
            float x0 = s0[j], x1 = s1[j], x2 = s2[j], x3 = s3[j];
            float mx = fmaxf(fmaxf(x0, x1), fmaxf(x2, x3));
#pragma unroll
            for (int msk = 8; msk >= 1; msk >>= 1)
                mx = fmaxf(mx, __shfl_xor(mx, msk, 16));
            mx = fmaxf(mrun[j], mx);
            const float corr = exp2f(mrun[j] - mx);
            mrun[j] = mx;
            const float p0 = exp2f(x0 - mx);
            const float p1 = exp2f(x1 - mx);
            const float p2 = exp2f(x2 - mx);
            const float p3 = exp2f(x3 - mx);
            o0[j] *= corr; o1[j] *= corr; o2[j] *= corr; o3[j] *= corr; o4[j] *= corr;

            _Float16* Pr = P + (j + hi * 8) * KP;
            Pr[lh]      = (_Float16)p0;
            Pr[16 + lh] = (_Float16)p1;
            Pr[32 + lh] = (_Float16)p2;
            Pr[48 + lh] = (_Float16)p3;
        }

        asm volatile("s_wait_dscnt 0" ::: "memory");       // wave-local P visibility

        const v16h pa0 = load_a16(P + lh * KP, hi);        // keys 0..31
        const v16h pa1 = load_a16(P + lh * KP + 32, hi);   // keys 32..63

        // ---- PV + ones-column row-sum ------------------------------------
        o0 = wmma16(pa0, load_b16(vt + (0 * 16 + lh) * KP +  0 + hi * 16), o0);
        o0 = wmma16(pa1, load_b16(vt + (0 * 16 + lh) * KP + 32 + hi * 16), o0);
        o1 = wmma16(pa0, load_b16(vt + (1 * 16 + lh) * KP +  0 + hi * 16), o1);
        o1 = wmma16(pa1, load_b16(vt + (1 * 16 + lh) * KP + 32 + hi * 16), o1);
        o2 = wmma16(pa0, load_b16(vt + (2 * 16 + lh) * KP +  0 + hi * 16), o2);
        o2 = wmma16(pa1, load_b16(vt + (2 * 16 + lh) * KP + 32 + hi * 16), o2);
        o3 = wmma16(pa0, load_b16(vt + (3 * 16 + lh) * KP +  0 + hi * 16), o3);
        o3 = wmma16(pa1, load_b16(vt + (3 * 16 + lh) * KP + 32 + hi * 16), o3);
        o4 = wmma16(pa0, vones, o4);
        o4 = wmma16(pa1, vones, o4);

        wait_async();
        __syncthreads();
    }

#pragma unroll
    for (int j = 0; j < 8; ++j) {
        const float inv = 1.0f / o4[j];
        const int m = q0 + j + hi * 8;
        float* dst = Out + (size_t)m * DIM + head * HD + lh;
        dst[0]  = o0[j] * inv;
        dst[16] = o1[j] * inv;
        dst[32] = o2[j] * inv;
        dst[48] = o3[j] * inv;
    }
}

// ------------------------------- launcher ----------------------------------

extern "C" void kernel_launch(void* const* d_in, const int* in_sizes, int n_in,
                              void* d_out, int out_size, void* d_ws, size_t ws_size,
                              hipStream_t stream) {
    const float* X  = (const float*)d_in[0];
    const float* Wq = (const float*)d_in[1];
    const float* bq = (const float*)d_in[2];
    const float* Wk = (const float*)d_in[3];
    const float* bk = (const float*)d_in[4];
    const float* Wv = (const float*)d_in[5];
    const float* bv = (const float*)d_in[6];
    float* out = (float*)d_out;

    char* ws = (char*)d_ws;
    const size_t SZ_X = (size_t)S_LEN * DIM * sizeof(_Float16);  // 8 MiB
    const size_t SZ_W = (size_t)DIM * DIM * sizeof(_Float16);    // 2 MiB
    _Float16* Xh  = (_Float16*)(ws);
    _Float16* Wtq = (_Float16*)(ws + SZ_X);
    _Float16* Wtk = Wtq + (size_t)DIM * DIM;
    _Float16* Wtv = Wtk + (size_t)DIM * DIM;
    _Float16* Qh  = (_Float16*)(ws + SZ_X + 3 * SZ_W);
    _Float16* Kh  = Qh + (size_t)S_LEN * DIM;
    _Float16* Vt  = Kh + (size_t)S_LEN * DIM;

    const float CSC = 0.125f * 1.4426950408889634f;  // 1/sqrt(64) * log2(e)

    const int n4 = (S_LEN * DIM) / 4;
    cvt_x_f16<<<dim3((n4 + 255) / 256), dim3(256), 0, stream>>>(X, Xh, n4);

    cvt_w_t<<<dim3(DIM / 256, DIM), dim3(256), 0, stream>>>(Wq, Wtq);
    cvt_w_t<<<dim3(DIM / 256, DIM), dim3(256), 0, stream>>>(Wk, Wtk);
    cvt_w_t<<<dim3(DIM / 256, DIM), dim3(256), 0, stream>>>(Wv, Wtv);

    dim3 ggrid(S_LEN / 128, DIM / 64);
    dim3 gblk(32, 4);
    gemm_bias_f16<<<ggrid, gblk, 0, stream>>>(Xh, Wtq, bq, Qh, CSC, 0);
    gemm_bias_f16<<<ggrid, gblk, 0, stream>>>(Xh, Wtk, bk, Kh, 1.0f, 0);
    gemm_bias_f16<<<ggrid, gblk, 0, stream>>>(Xh, Wtv, bv, Vt, 1.0f, 1);

    attn_fa<<<dim3(S_LEN / 64, NH), dim3(32, 4), 0, stream>>>(Qh, Kh, Vt, out);
}
